// CrossAttention_84035330113965
// MI455X (gfx1250) — matrix operand order repaired and checked by reading, not compiled, not run on hardware
//
#include <hip/hip_runtime.h>
#include <hip/hip_bf16.h>

#define C_DIM 256
#define N_DIM 4096
#define B_DIM 4

typedef __attribute__((ext_vector_type(16))) __bf16 v16bf;
typedef __attribute__((ext_vector_type(8)))  __bf16 v8bf;
typedef __attribute__((ext_vector_type(8)))  float  v8f;
typedef __attribute__((ext_vector_type(4)))  float  v4f;
typedef int v4i_vs __attribute__((vector_size(16)));   // matches builtin param type

union V16U { v16bf v; v8bf h[2]; };
union V8FU { v8f v; v4f q[2]; float f[8]; };

// Probe for the CDNA5 async global->LDS copy builtin (ASYNCcnt path).
#if defined(__has_builtin)
# if __has_builtin(__builtin_amdgcn_global_load_async_to_lds_b128)
#  define HAVE_ASYNC_LDS 1
# endif
#endif
#ifndef HAVE_ASYNC_LDS
# define HAVE_ASYNC_LDS 0
#endif

#if HAVE_ASYNC_LDS
typedef __attribute__((address_space(1))) v4i_vs* g_v4i_ptr;
typedef __attribute__((address_space(3))) v4i_vs* l_v4i_ptr;
#endif

// Copy 16 bytes global -> LDS. Async (no VGPR round-trip) when available.
__device__ __forceinline__ void copy16_g2l(const __bf16* g, __bf16* l) {
#if HAVE_ASYNC_LDS
    __builtin_amdgcn_global_load_async_to_lds_b128(
        (g_v4i_ptr)(g), (l_v4i_ptr)(l), 0, 0);
#else
    *(v8bf*)l = *(const v8bf*)g;
#endif
}

__device__ __forceinline__ void g2l_wait() {
#if HAVE_ASYNC_LDS
# if __has_builtin(__builtin_amdgcn_s_wait_asynccnt)
    __builtin_amdgcn_s_wait_asynccnt(0);
# else
    asm volatile("s_wait_asynccnt 0x0" ::: "memory");
# endif
#endif
}

// A-fragment (16x32 bf16, M x K): lane row M = lane%16; lane's 16 halves are
// K = base..base+7 and base+16..base+23, base = (lane>>4)*8.
__device__ __forceinline__ v16bf load_a_bf16(const __bf16* row, int base) {
    V16U u;
    u.h[0] = *(const v8bf*)(row + base);
    u.h[1] = *(const v8bf*)(row + base + 16);
    return u.v;
}

// Same A layout, converting from f32 source (weights).
__device__ __forceinline__ v16bf load_a_f32(const float* row, int base) {
    V16U u;
#pragma unroll
    for (int i = 0; i < 8; ++i) u.v[i]     = (__bf16)row[base + i];
#pragma unroll
    for (int i = 0; i < 8; ++i) u.v[8 + i] = (__bf16)row[base + 16 + i];
    return u.v;
}

// B-fragment (32x16 bf16, K x N): lane column N = lane%16,
// K = base..base+7 and base+16..base+23, base = (lane>>4)*8.
__device__ __forceinline__ v16bf load_b_bf16(const __bf16* col, int base) {
    V16U u;
    u.h[0] = *(const v8bf*)(col + base);
    u.h[1] = *(const v8bf*)(col + base + 16);
    return u.v;
}

__device__ __forceinline__ v8f wmma_bf16(v16bf a, v16bf b, v8f c) {
    return __builtin_amdgcn_wmma_f32_16x16x32_bf16(false, a, false, b, (short)0, c, false, false);
}

// ---------------------------------------------------------------------------
// Projection: Y[o,n] = relu(scale[o]*(sum_c W[o,c] X[c,n]) + bias[o])
// TOUT=true : output bf16 [B][N][C] (pixel-major, for Q/K A- and B-fragments)
// TOUT=false: output bf16 [B][C][N] (channel-major, for V B-fragments)
// Block: 256 threads (8 waves); per block 16 pixels x all 256 out channels.
// ---------------------------------------------------------------------------
template <bool TOUT>
__global__ __launch_bounds__(256) void proj_kernel(
    const float* __restrict__ X, const float* __restrict__ W,
    const float* __restrict__ scale, const float* __restrict__ bias,
    __bf16* __restrict__ Outp)
{
    __shared__ __attribute__((aligned(16))) __bf16 Xsh[16 * C_DIM]; // [n][c]

    const int b   = blockIdx.y;
    const int n0  = blockIdx.x * 16;
    const int tid = threadIdx.x;
    const float* Xb = X + (size_t)b * C_DIM * N_DIM;

    // Stage X tile [16 n][256 c] into LDS with f32->bf16 convert (transpose).
    {
        const int c = tid; // 0..255
        const float* src = Xb + (size_t)c * N_DIM + n0;
#pragma unroll
        for (int j = 0; j < 16; ++j)
            Xsh[j * C_DIM + c] = (__bf16)src[j];
    }
    __syncthreads();

    const int lane = tid & 31;
    const int wave = tid >> 5;
    const int l16  = lane & 15;
    const int hi   = lane >> 4;
    const int aoff = hi << 3;
    const int boff = hi << 3;

#pragma unroll
    for (int ph = 0; ph < 2; ++ph) {
        const int ot = wave + ph * 8;            // output-channel tile (16 o)
        v8f acc = {};
        const float*  wrow = W + (size_t)(ot * 16 + l16) * C_DIM;
        const __bf16* bcol = Xsh + l16 * C_DIM;
#pragma unroll
        for (int kk = 0; kk < 8; ++kk) {         // K = 256 = 8 x 32
            v16bf a  = load_a_f32(wrow, kk * 32 + aoff);
            v16bf bm = load_b_bf16(bcol, kk * 32 + boff);
            acc = wmma_bf16(a, bm, acc);
        }
        // Epilogue: BN affine + ReLU. C/D layout: VGPR i -> M = i + hi*8.
        const int obase = ot * 16 + hi * 8;
        const int n = n0 + l16;
        __bf16 y[8];
#pragma unroll
        for (int i = 0; i < 8; ++i) {
            const int o = obase + i;
            float t = acc[i] * scale[o] + bias[o];
            y[i] = (__bf16)(t > 0.f ? t : 0.f);
        }
        if (TOUT) {
            v8bf r;
#pragma unroll
            for (int i = 0; i < 8; ++i) r[i] = y[i];
            *(v8bf*)(Outp + ((size_t)b * N_DIM + n) * C_DIM + obase) = r;
        } else {
#pragma unroll
            for (int i = 0; i < 8; ++i)
                Outp[((size_t)b * C_DIM + obase + i) * N_DIM + n] = y[i];
        }
    }
}

// ---------------------------------------------------------------------------
// Flash attention: per wave 16 queries, keys streamed in blocks of 32.
// Q A-fragments pinned in VGPRs; K [key][c] and V [c][key] tiles staged in
// LDS (async global->LDS when available); online softmax via xor-shuffles.
// Block: 256 threads = 8 waves = 128 queries.
// ---------------------------------------------------------------------------
__global__ __launch_bounds__(256) void attn_kernel(
    const __bf16* __restrict__ QT, const __bf16* __restrict__ KT,
    const __bf16* __restrict__ VC, float* __restrict__ Out)
{
    __shared__ __attribute__((aligned(16))) __bf16 Ksh[32 * C_DIM];  // [key][c]
    __shared__ __attribute__((aligned(16))) __bf16 Vsh[C_DIM * 32];  // [c][key]
    __shared__ __attribute__((aligned(16))) __bf16 Psh[8 * 16 * 32]; // per-wave [q][k]

    const int b    = blockIdx.y;
    const int tid  = threadIdx.x;
    const int lane = tid & 31;
    const int wave = tid >> 5;
    const int l16  = lane & 15;
    const int hi   = lane >> 4;
    const int qbase = blockIdx.x * 128 + wave * 16;

    const __bf16* Qb = QT + (size_t)b * N_DIM * C_DIM;
    const __bf16* Kb = KT + (size_t)b * N_DIM * C_DIM;
    const __bf16* Vb = VC + (size_t)b * C_DIM * N_DIM;   // channel-major

    // This wave's Q A-fragments (16 queries x 256 channels) stay in VGPRs.
    v16bf qf[8];
    {
        const __bf16* qrow = Qb + (size_t)(qbase + l16) * C_DIM;
#pragma unroll
        for (int kk = 0; kk < 8; ++kk)
            qf[kk] = load_a_bf16(qrow, kk * 32 + (hi << 3));
    }

    v8f zero = {};
    v8f O[16];
#pragma unroll
    for (int i = 0; i < 16; ++i) O[i] = zero;
    float mrow[8], lrow[8];
#pragma unroll
    for (int r = 0; r < 8; ++r) { mrow[r] = -1e30f; lrow[r] = 0.f; }

    __bf16* Pw = Psh + wave * (16 * 32);
    const float sc = 0.0625f;        // C^-0.5 = 1/16

    for (int kb = 0; kb < N_DIM; kb += 32) {
        __syncthreads();             // previous tile fully consumed
        // Stage K tile [32 key][256 c]: 64B per thread, contiguous.
        {
            const int key = tid >> 3;
            const int off = (tid & 7) * 32;
            const __bf16* s = Kb + (size_t)(kb + key) * C_DIM + off;
            __bf16* d = Ksh + key * C_DIM + off;
#pragma unroll
            for (int ch = 0; ch < 4; ++ch) copy16_g2l(s + ch * 8, d + ch * 8);
            if (kb + 32 < N_DIM)     // prefetch next K tile -> global_prefetch_b8
                __builtin_prefetch(Kb + (size_t)(kb + 32 + key) * C_DIM + off, 0, 1);
        }
        // Stage V tile [256 c][32 key]: V is channel-major, straight copy.
        {
            const __bf16* s = Vb + (size_t)tid * N_DIM + kb;
            __bf16* d = Vsh + tid * 32;
#pragma unroll
            for (int ch = 0; ch < 4; ++ch) copy16_g2l(s + ch * 8, d + ch * 8);
        }
        g2l_wait();
        __syncthreads();

        // S = Q^T K for 2 key sub-tiles (16 keys each), K-dim = 256.
        v8f s0 = {}, s1 = {};
#pragma unroll
        for (int kk = 0; kk < 8; ++kk) {
            v16bf b0 = load_b_bf16(Ksh + l16 * C_DIM, kk * 32 + (hi << 3));
            s0 = wmma_bf16(qf[kk], b0, s0);
        }
#pragma unroll
        for (int kk = 0; kk < 8; ++kk) {
            v16bf b1 = load_b_bf16(Ksh + (16 + l16) * C_DIM, kk * 32 + (hi << 3));
            s1 = wmma_bf16(qf[kk], b1, s1);
        }

        // Online softmax: row m = r + hi*8 lives in one 16-lane half, so
        // xor-1/2/4/8 shuffles reduce exactly over that row's 16 key columns.
        float fscale[8];
#pragma unroll
        for (int r = 0; r < 8; ++r) {
            float a0 = s0[r] * sc, a1 = s1[r] * sc;
            float mx = fmaxf(a0, a1);
            mx = fmaxf(mx, __shfl_xor(mx, 1));
            mx = fmaxf(mx, __shfl_xor(mx, 2));
            mx = fmaxf(mx, __shfl_xor(mx, 4));
            mx = fmaxf(mx, __shfl_xor(mx, 8));
            const float mnew = fmaxf(mrow[r], mx);
            const float f  = __expf(mrow[r] - mnew);
            const float p0 = __expf(a0 - mnew);
            const float p1 = __expf(a1 - mnew);
            float rs = p0 + p1;
            rs += __shfl_xor(rs, 1);
            rs += __shfl_xor(rs, 2);
            rs += __shfl_xor(rs, 4);
            rs += __shfl_xor(rs, 8);
            lrow[r] = lrow[r] * f + rs;
            mrow[r] = mnew;
            fscale[r] = f;
            const int m = r + hi * 8;            // P -> LDS in A-matrix [q][k] order
            Pw[m * 32 + l16]      = (__bf16)p0;
            Pw[m * 32 + 16 + l16] = (__bf16)p1;
        }
        // Rescale running context by exp(m_old - m_new).
#pragma unroll
        for (int ct = 0; ct < 16; ++ct) {
#pragma unroll
            for (int r = 0; r < 8; ++r)
                O[ct][r] = O[ct][r] * fscale[r];
        }
        // ctx += P (16x32) x V^T (32 x 16 per channel tile).
        v16bf pa = load_a_bf16(Pw + l16 * 32, hi << 3);
#pragma unroll
        for (int ct = 0; ct < 16; ++ct) {
            v16bf bv = load_b_bf16(Vsh + (ct * 16 + l16) * 32, hi << 3);
            O[ct] = wmma_bf16(pa, bv, O[ct]);
        }
    }

    // Epilogue: divide by softmax denominator, store f32 [B][C][N].
    float inv[8];
#pragma unroll
    for (int r = 0; r < 8; ++r) inv[r] = 1.0f / lrow[r];
    float* Ob = Out + (size_t)b * C_DIM * N_DIM;
#pragma unroll
    for (int ct = 0; ct < 16; ++ct) {
        const int c = ct * 16 + l16;
        V8FU u;
#pragma unroll
        for (int r = 0; r < 8; ++r) u.f[r] = O[ct][r] * inv[r];
        v4f* dst = (v4f*)(Ob + (size_t)c * N_DIM + qbase + hi * 8);
        dst[0] = u.q[0];
        dst[1] = u.q[1];
    }
}

extern "C" void kernel_launch(void* const* d_in, const int* in_sizes, int n_in,
                              void* d_out, int out_size, void* d_ws, size_t ws_size,
                              hipStream_t stream) {
    (void)in_sizes; (void)n_in; (void)out_size; (void)ws_size;
    const float* qfeat = (const float*)d_in[0];
    const float* kfeat = (const float*)d_in[1];
    const float* Wq = (const float*)d_in[2];
    const float* Wk = (const float*)d_in[3];
    const float* Wv = (const float*)d_in[4];
    const float* sq = (const float*)d_in[5];
    const float* bq = (const float*)d_in[6];
    const float* sk = (const float*)d_in[7];
    const float* bk = (const float*)d_in[8];
    const float* sv = (const float*)d_in[9];
    const float* bvp = (const float*)d_in[10];

    const size_t tile_bytes = (size_t)B_DIM * N_DIM * C_DIM * sizeof(__bf16); // 8 MB
    char* ws = (char*)d_ws;
    __bf16* QT = (__bf16*)(ws);                   // [B][N][C]
    __bf16* KT = (__bf16*)(ws + tile_bytes);      // [B][N][C]
    __bf16* VC = (__bf16*)(ws + 2 * tile_bytes);  // [B][C][N]

    dim3 pg(N_DIM / 16, B_DIM);
    proj_kernel<true ><<<pg, dim3(256), 0, stream>>>(qfeat, Wq, sq, bq, QT);
    proj_kernel<true ><<<pg, dim3(256), 0, stream>>>(kfeat, Wk, sk, bk, KT);
    proj_kernel<false><<<pg, dim3(256), 0, stream>>>(kfeat, Wv, sv, bvp, VC);

    dim3 ag(N_DIM / 128, B_DIM);
    attn_kernel<<<ag, dim3(256), 0, stream>>>(QT, KT, VC, (float*)d_out);
}
